// MultiHeadedAttention_20280835572097
// MI455X (gfx1250) — matrix operand order repaired and checked by reading, not compiled, not run on hardware
//
#include <hip/hip_runtime.h>
#include <hip/hip_bf16.h>

typedef __attribute__((ext_vector_type(16))) __bf16 v16bf;
typedef __attribute__((ext_vector_type(8)))  float  v8f;
typedef __attribute__((ext_vector_type(4)))  unsigned u32x4;
typedef __attribute__((ext_vector_type(8)))  int      i32x8;
typedef __attribute__((ext_vector_type(4)))  int      i32x4;

static constexpr int HEADS  = 8;
static constexpr int DMODEL = 512;
static constexpr int DK     = 64;
static constexpr int SEQ    = 2048;
static constexpr int BATCH  = 4;
static constexpr int QTILE  = 128;   // 8 waves x 16 q-rows
static constexpr int KTILE  = 64;    // keys per TDM tile

__device__ __forceinline__ unsigned f2bf1(float f) {
  unsigned u = __float_as_uint(f);
  return (u + 0x7FFFu + ((u >> 16) & 1u)) >> 16;   // round-to-nearest-even bf16
}
__device__ __forceinline__ unsigned pack2bf(float a, float b) {
  return f2bf1(a) | (f2bf1(b) << 16);
}

union B16 { v16bf v; unsigned u[8]; };

// ---------------------------------------------------------------------------
// TDM: DMA one 64x64 bf16 tile (row pitch = stride elements) global -> LDS.
// D# built per CDNA5 ISA bitfields; 2-D tile so
// groups 2/3 are zero. Issue under a wave-uniform guard; TENSORcnt tracks it.
// ---------------------------------------------------------------------------
__device__ __forceinline__ void tdm_load_64x64(unsigned lds_byte_addr,
                                               const unsigned short* gptr,
                                               int stride_elems) {
  u32x4 g0;
  unsigned long long ga = (unsigned long long)gptr;
  g0[0] = 1u;                                               // count=1 (valid user D#)
  g0[1] = lds_byte_addr;                                    // LDS dest (bytes)
  g0[2] = (unsigned)ga;                                     // global_addr[31:0]
  g0[3] = (unsigned)((ga >> 32) & 0x1FFFFFFull) | (2u << 30); // addr[56:32] | type=2
  i32x8 g1;
  g1[0] = 1 << 16;            // workgroup_mask=0, data_size=1 (2 bytes)
  g1[1] = 64 << 16;           // tensor_dim0 = 64  (bits 63:48)
  g1[2] = 64 << 16;           // tensor_dim1 = 64  (bits 111:80, low half)
  g1[3] = 64 << 16;           // tile_dim0   = 64  (bits 127:112)
  g1[4] = 64;                 // tile_dim1   = 64, tile_dim2 = 0
  g1[5] = stride_elems;       // tensor_dim0_stride[31:0]
  g1[6] = 0;
  g1[7] = 0;
  i32x4 z4 = {};
#if defined(__clang_major__) && __clang_major__ >= 23
  i32x8 z8 = {};
  __builtin_amdgcn_tensor_load_to_lds(g0, g1, z4, z4, z8, 0);
#else
  __builtin_amdgcn_tensor_load_to_lds(g0, g1, z4, z4, 0);
#endif
}

__device__ __forceinline__ unsigned lds_off(const void* p) {
  // generic (flat) shared pointer: low 32 bits are the LDS byte offset
  return (unsigned)(unsigned long long)p;
}

// ---------------------------------------------------------------------------
// Kernel 0 (prep): K -> bf16 (same layout), V -> bf16 transposed per (n,h):
// vt[n*H+h][d=64][S=2048]. Makes the attention kernel a pure TDM+WMMA pipe.
// ---------------------------------------------------------------------------
__global__ __launch_bounds__(256) void prep_kernel(
    const float* __restrict__ kg, const float* __restrict__ vg,
    unsigned short* __restrict__ kbf, unsigned short* __restrict__ vt) {
  __shared__ unsigned short t_v[DK][KTILE];   // transposed V tile, 8 KB

  const int n   = blockIdx.y >> 3;
  const int h   = blockIdx.y & 7;
  const int k0  = blockIdx.x * KTILE;
  const int tid = threadIdx.x;
  const int row = tid >> 2;        // 0..63 key row
  const int seg = tid & 3;         // 16 d each

  const size_t gbase = ((size_t)(n * SEQ + k0 + row)) * DMODEL + h * DK + seg * 16;
  // K: elementwise convert, coalesced in/out
  {
    const float* src = kg + gbase;
    unsigned tmp[8];
    #pragma unroll
    for (int i = 0; i < 4; ++i) {
      float4 f = *(const float4*)(src + i * 4);
      tmp[i * 2 + 0] = pack2bf(f.x, f.y);
      tmp[i * 2 + 1] = pack2bf(f.z, f.w);
    }
    uint4* dst = (uint4*)(kbf + gbase);
    dst[0] = make_uint4(tmp[0], tmp[1], tmp[2], tmp[3]);
    dst[1] = make_uint4(tmp[4], tmp[5], tmp[6], tmp[7]);
  }
  // V: convert + transpose through LDS
  {
    const float* src = vg + gbase;
    #pragma unroll
    for (int i = 0; i < 4; ++i) {
      float4 f = *(const float4*)(src + i * 4);
      int dd = seg * 16 + i * 4;
      t_v[dd + 0][row] = (unsigned short)f2bf1(f.x);
      t_v[dd + 1][row] = (unsigned short)f2bf1(f.y);
      t_v[dd + 2][row] = (unsigned short)f2bf1(f.z);
      t_v[dd + 3][row] = (unsigned short)f2bf1(f.w);
    }
  }
  __syncthreads();
  {
    const int d  = tid >> 2;          // 0..63
    const int ks = (tid & 3) << 4;    // 16 keys
    unsigned short* dst = vt + ((size_t)(n * HEADS + h) * DK + d) * SEQ + k0 + ks;
    *(uint4*)(dst + 0) = *(const uint4*)&t_v[d][ks + 0];
    *(uint4*)(dst + 8) = *(const uint4*)&t_v[d][ks + 8];
  }
}

// ---------------------------------------------------------------------------
// Kernel 1: flash attention. TDM double-buffers K/V^T tiles into LDS while
// WMMAs run; S^T = K @ Q^T keeps softmax lane-local and feeds P@V in-lane.
// ---------------------------------------------------------------------------
__global__ __launch_bounds__(256) void attn_kernel(
    const float* __restrict__ qg, const unsigned short* __restrict__ kbf,
    const unsigned short* __restrict__ vt, unsigned short* __restrict__ attn16) {
  __shared__ __align__(16) unsigned short lds_k[2][KTILE][DK];   // 2 x 8 KB
  __shared__ __align__(16) unsigned short lds_vt[2][DK][KTILE];  // 2 x 8 KB

  const int n    = blockIdx.y >> 3;
  const int h    = blockIdx.y & 7;
  const int q0   = blockIdx.x * QTILE;
  const int tid  = threadIdx.x;
  const int wave = tid >> 5;
  const int lane = tid & 31;
  const int ln16 = lane & 15;
  const int half = lane >> 4;

  const int   qrow  = q0 + wave * 16 + ln16;
  const float SCALE = 0.044194173824159216f;   // 1/sqrt(512) (reference quirk)

  // Q in B-operand layout (B = Q^T): vgpr v holds d = c*32 + (v&3)*2 + (v>>2)*16 + half*8
  B16 qb[2];
  {
    const float* qbase = qg + ((size_t)(n * SEQ + qrow)) * DMODEL + h * DK;
    #pragma unroll
    for (int c = 0; c < 2; ++c) {
      #pragma unroll
      for (int vv = 0; vv < 8; ++vv) {
        int dd = c * 32 + (vv & 3) * 2 + (vv >> 2) * 16 + half * 8;
        float2 f = *(const float2*)(qbase + dd);
        qb[c].u[vv] = pack2bf(f.x, f.y);
      }
    }
  }

  float m = -__builtin_inff();
  float l = 0.0f;
  v8f   o[4] = {};

  const unsigned short* kbase = kbf + ((size_t)n * SEQ) * DMODEL + h * DK;
  const unsigned short* vbase = vt + ((size_t)(n * HEADS + h) * DK) * SEQ;
  const int NT = SEQ / KTILE;

  if (wave == 0) {   // prologue: DMA tile 0
    tdm_load_64x64(lds_off(&lds_k[0][0][0]),  kbase, DMODEL);
    tdm_load_64x64(lds_off(&lds_vt[0][0][0]), vbase, SEQ);
  }

  for (int it = 0; it < NT; ++it) {
    const int buf = it & 1;
    if (wave == 0) {
      if (it + 1 < NT) {   // DMA next tile into the other buffer, then wait tile `it`
        tdm_load_64x64(lds_off(&lds_k[buf ^ 1][0][0]),
                       kbase + (size_t)(it + 1) * KTILE * DMODEL, DMODEL);
        tdm_load_64x64(lds_off(&lds_vt[buf ^ 1][0][0]),
                       vbase + (it + 1) * KTILE, SEQ);
        __builtin_amdgcn_s_wait_tensorcnt((short)2);
      } else {
        __builtin_amdgcn_s_wait_tensorcnt((short)0);
      }
    }
    __syncthreads();

    #pragma unroll
    for (int c = 0; c < KTILE; c += 32) {
      // ---- S^T tiles (keys x q): two 16-key subtiles, K-dim = d (2 chunks) ----
      v8f s[2] = {};
      #pragma unroll
      for (int tt = 0; tt < 2; ++tt) {
        int keyr = c + tt * 16 + ln16;
        #pragma unroll
        for (int dc = 0; dc < 2; ++dc) {
          B16 ka;   // A layout: d = dc*32 + (v&3)*2 + (v>>2)*16 + half*8
          #pragma unroll
          for (int vv = 0; vv < 8; ++vv) {
            int dd = dc * 32 + (vv & 3) * 2 + (vv >> 2) * 16 + half * 8;
            ka.u[vv] = *(const unsigned*)&lds_k[buf][keyr][dd];
          }
          s[tt] = __builtin_amdgcn_wmma_f32_16x16x32_bf16(
              false, ka.v, false, qb[dc].v, (short)0, s[tt], false, false);
        }
      }

      // ---- online softmax over keys (lane-local + one half-swap) ----
      float mx = -__builtin_inff();
      #pragma unroll
      for (int r = 0; r < 8; ++r) mx = fmaxf(mx, fmaxf(s[0][r], s[1][r]));
      mx *= SCALE;
      mx = fmaxf(mx, __shfl_xor(mx, 16, 32));
      float mnew = fmaxf(m, mx);
      float cf   = __expf(m - mnew);
      float p[2][8];
      float sum = 0.0f;
      #pragma unroll
      for (int tt = 0; tt < 2; ++tt)
        #pragma unroll
        for (int r = 0; r < 8; ++r) {
          p[tt][r] = __expf(s[tt][r] * SCALE - mnew);
          sum += p[tt][r];
        }
      sum += __shfl_xor(sum, 16, 32);
      l = l * cf + sum;
      m = mnew;
      #pragma unroll
      for (int dt = 0; dt < 4; ++dt)
        #pragma unroll
        for (int r = 0; r < 8; ++r) o[dt][r] *= cf;

      // ---- P: C-layout of S^T -> A-layout by pure in-lane packing ----
      B16 pa;
      #pragma unroll
      for (int vv = 0; vv < 8; ++vv)
        pa.u[vv] = pack2bf(p[vv >> 2][2 * (vv & 3)], p[vv >> 2][2 * (vv & 3) + 1]);

      // ---- O += P @ V ----
      #pragma unroll
      for (int dt = 0; dt < 4; ++dt) {
        B16 vbm;  // B layout: key = c + (v&3)*2 + (v>>2)*16 + half*8 (pairs contiguous in V^T)
        int dd = dt * 16 + ln16;
        #pragma unroll
        for (int vv = 0; vv < 8; ++vv) {
          int key = c + (vv & 3) * 2 + (vv >> 2) * 16 + half * 8;
          vbm.u[vv] = *(const unsigned*)&lds_vt[buf][dd][key];
        }
        o[dt] = __builtin_amdgcn_wmma_f32_16x16x32_bf16(
            false, pa.v, false, vbm.v, (short)0, o[dt], false, false);
      }
    }
    __syncthreads();   // all waves done reading `buf`; safe to DMA over it
  }

  // ---- normalize (l lives in lane==q; broadcast to output rows), store bf16 ----
  float invl[8];
  #pragma unroll
  for (int r = 0; r < 8; ++r) {
    float lr = __shfl(l, r + half * 8, 32);
    invl[r] = 1.0f / lr;
  }
  #pragma unroll
  for (int dt = 0; dt < 4; ++dt) {
    #pragma unroll
    for (int r = 0; r < 8; ++r) {
      int row = q0 + wave * 16 + r + half * 8;
      attn16[((size_t)(n * SEQ + row)) * DMODEL + h * DK + dt * 16 + ln16] =
          (unsigned short)f2bf1(o[dt][r] * invl[r]);
    }
  }
}

// ---------------------------------------------------------------------------
// Kernel 2: out = attn @ W^T + b (8192x512x512). X already bf16 -> raw b128
// copies into LDS; only W is converted. 128x64 block tile, WMMA bf16.
// ---------------------------------------------------------------------------
__global__ __launch_bounds__(256) void proj_kernel(
    const unsigned short* __restrict__ x16, const float* __restrict__ w,
    const float* __restrict__ b, float* __restrict__ out) {
  __shared__ __align__(16) unsigned short xs[128][32];   // [m][k] 8 KB
  __shared__ __align__(16) unsigned short wsd[64][32];   // [n][k] 4 KB

  const int m0   = blockIdx.y * 128;
  const int n0   = blockIdx.x * 64;
  const int tid  = threadIdx.x;
  const int wave = tid >> 5;
  const int lane = tid & 31;
  const int ln16 = lane & 15;
  const int half = lane >> 4;

  v8f acc[4] = {};

  const int xrow = tid >> 1;   // 0..127
  const int xseg = tid & 1;
  const int wrow = tid >> 2;   // 0..63
  const int wseg = tid & 3;

  for (int k0 = 0; k0 < DMODEL; k0 += 32) {
    const unsigned short* xb = x16 + (size_t)(m0 + xrow) * DMODEL + k0 + xseg * 16;
    if (k0 + 32 < DMODEL) __builtin_prefetch(xb + 32, 0, 0);  // global_prefetch_b8
    *(uint4*)&xs[xrow][xseg * 16 + 0] = *(const uint4*)(xb + 0);
    *(uint4*)&xs[xrow][xseg * 16 + 8] = *(const uint4*)(xb + 8);

    const float* wb = w + ((size_t)(n0 + wrow)) * DMODEL + k0 + wseg * 8;
    #pragma unroll
    for (int i = 0; i < 2; ++i) {
      float4 f = *(const float4*)(wb + i * 4);
      *(unsigned*)&wsd[wrow][wseg * 8 + i * 4 + 0] = pack2bf(f.x, f.y);
      *(unsigned*)&wsd[wrow][wseg * 8 + i * 4 + 2] = pack2bf(f.z, f.w);
    }
    __syncthreads();

    B16 xa;   // A layout
    #pragma unroll
    for (int vv = 0; vv < 8; ++vv) {
      int kk = (vv & 3) * 2 + (vv >> 2) * 16 + half * 8;
      xa.u[vv] = *(const unsigned*)&xs[wave * 16 + ln16][kk];
    }
    #pragma unroll
    for (int dt = 0; dt < 4; ++dt) {
      B16 wv;   // B layout: k = (v&3)*2 + (v>>2)*16 + half*8
      #pragma unroll
      for (int vv = 0; vv < 8; ++vv)
        wv.u[vv] = *(const unsigned*)&wsd[dt * 16 + ln16][(vv & 3) * 2 + (vv >> 2) * 16 + half * 8];
      acc[dt] = __builtin_amdgcn_wmma_f32_16x16x32_bf16(
          false, xa.v, false, wv.v, (short)0, acc[dt], false, false);
    }
    __syncthreads();
  }

  #pragma unroll
  for (int dt = 0; dt < 4; ++dt) {
    int j = n0 + dt * 16 + ln16;
    float bb = b[j];
    #pragma unroll
    for (int r = 0; r < 8; ++r) {
      int row = m0 + wave * 16 + r + half * 8;
      out[(size_t)row * DMODEL + j] = acc[dt][r] + bb;
    }
  }
}

extern "C" void kernel_launch(void* const* d_in, const int* in_sizes, int n_in,
                              void* d_out, int out_size, void* d_ws, size_t ws_size,
                              hipStream_t stream) {
  const float* q = (const float*)d_in[0];
  const float* k = (const float*)d_in[1];
  const float* v = (const float*)d_in[2];
  const float* W = (const float*)d_in[3];
  const float* b = (const float*)d_in[4];
  float* out = (float*)d_out;

  // scratch: kbf (8MB) | vt (8MB) | attn16 (8MB), all bf16
  unsigned short* kbf    = (unsigned short*)d_ws;
  unsigned short* vt     = kbf + (size_t)BATCH * SEQ * DMODEL;
  unsigned short* attn16 = vt + (size_t)BATCH * HEADS * DK * SEQ;

  dim3 g0(SEQ / KTILE, BATCH * HEADS);        // (32, 32)
  prep_kernel<<<g0, 256, 0, stream>>>(k, v, kbf, vt);

  dim3 g1(SEQ / QTILE, BATCH * HEADS);        // (16, 32)
  attn_kernel<<<g1, 256, 0, stream>>>(q, kbf, vt, attn16);

  dim3 g2(DMODEL / 64, (BATCH * SEQ) / 128);  // (8, 64)
  proj_kernel<<<g2, 256, 0, stream>>>(attn16, W, b, out);
}
